// Encoder_GRU_1_33062658245300
// MI455X (gfx1250) — compile-verified, matrix operand order checked by reading, not run on hardware
//
#include <hip/hip_runtime.h>
#include <hip/hip_bf16.h>

#define B_  512
#define T_  64
#define E_  256
#define H_  512
#define G3_ (3 * H_)

typedef __attribute__((ext_vector_type(16))) __bf16 v16bf;
typedef __attribute__((ext_vector_type(8)))  float  v8f;

union ABu { v16bf v; uint4 q[2]; };

// Load one 16x32 (or 32x16) bf16 WMMA fragment row for this lane.
// ISA layout (CDNA5 §7.12.2): lanes 0-15 hold K=0..7 and 16..23,
// lanes 16-31 hold K=8..15 and 24..31 of their row.
__device__ __forceinline__ v16bf load_ab32(const __bf16* __restrict__ p, int lane) {
  const int k0 = (lane & 16) ? 8 : 0;
  ABu u;
  u.q[0] = *(const uint4*)(p + k0);
  u.q[1] = *(const uint4*)(p + k0 + 16);
  return u.v;
}

__device__ __forceinline__ v8f wmma_bf16(v16bf a, v16bf b, v8f c) {
  return __builtin_amdgcn_wmma_f32_16x16x32_bf16(
      /*neg_a=*/false, a, /*neg_b=*/false, b,
      /*c_mod=*/(short)0, c, /*reuse_a=*/false, /*reuse_b=*/false);
}

// ---------------- prep kernels ----------------

__global__ void init_state(float* __restrict__ h32, __bf16* __restrict__ h16, int n) {
  int i = blockIdx.x * blockDim.x + threadIdx.x;
  if (i < n) { h32[i] = 0.0f; h16[i] = (__bf16)0.0f; }
}

// Gather embedding rows, cast to bf16, emit mask (as float 0/1).
__global__ void embed_cast(const int* __restrict__ src_ids, const float* __restrict__ emb,
                           __bf16* __restrict__ xbf, float* __restrict__ out_mask) {
  const int bt = blockIdx.x;           // b*T + t
  const int id = src_ids[bt];
  const int e  = threadIdx.x;          // 0..E_-1
  xbf[(size_t)bt * E_ + e] = (__bf16)emb[(size_t)id * E_ + e];
  if (e == 0) out_mask[bt] = (id != 0) ? 1.0f : 0.0f;
}

// Cast f32 weight [K, 3H] -> bf16 transposed [3H, K] (K-contiguous per output column)
__global__ void wcast_t(const float* __restrict__ src, __bf16* __restrict__ dst, int K) {
  const int n = blockIdx.x;            // 0..G3_-1
  for (int k = threadIdx.x; k < K; k += blockDim.x)
    dst[(size_t)n * K + k] = (__bf16)src[(size_t)k * G3_ + n];
}

// ---------------- per-timestep fused GRU kernel ----------------
// One wave per 16x16 output tile of h_new. 2048 waves: dir(2) x Mtiles(32) x Ntiles(32).
// Each wave accumulates four GEMM fragments with v_wmma_f32_16x16x32_bf16:
//   Z  = x_t@Wz + h@Uz   (K = 256 + 512)
//   R  = x_t@Wr + h@Ur   (K = 256 + 512)
//   XH = x_t@Wh          (K = 256)
//   RH = h@Uh            (K = 512)      (kept separate: reset_after=True)
__global__ __launch_bounds__(256) void gru_step(
    const __bf16* __restrict__ xbf,
    const __bf16* __restrict__ Wt_f, const __bf16* __restrict__ Ut_f,
    const __bf16* __restrict__ Wt_b, const __bf16* __restrict__ Ut_b,
    const float* __restrict__ bias_f, const float* __restrict__ bias_b,
    const int* __restrict__ src_ids,
    const float* __restrict__ h32_in, float* __restrict__ h32_out,
    const __bf16* __restrict__ h16_in, __bf16* __restrict__ h16_out,
    float* __restrict__ out_seq, float* __restrict__ out_hid, int t)
{
  const int lane   = threadIdx.x & 31;
  const int waveId = blockIdx.x * (blockDim.x >> 5) + (threadIdx.x >> 5);
  const int dir    = waveId >> 10;          // uniform per block (128 blocks/dir)
  const int rem    = waveId & 1023;
  const int mi     = rem >> 5;              // 0..31 over B
  const int ni     = rem & 31;              // 0..31 over H
  const int t_eff  = dir ? (T_ - 1 - t) : t;

  const __bf16* Wt = dir ? Wt_b : Wt_f;
  const __bf16* Ut = dir ? Ut_b : Ut_f;
  const float*  bb = dir ? bias_b : bias_f; // [0..3H) = b_in, [3H..6H) = b_rec

  const int mbase = mi * 16;
  const int nbase = ni * 16;
  const int arow  = mbase + (lane & 15);    // A-matrix row (M) for this lane
  const int bcol  = nbase + (lane & 15);    // B-matrix column (N) for this lane

  const __bf16* axrow = xbf + ((size_t)arow * T_ + t_eff) * E_;
  const __bf16* ahrow = h16_in + (size_t)dir * B_ * H_ + (size_t)arow * H_;

  const __bf16* wz = Wt + (size_t)(0 * H_ + bcol) * E_;
  const __bf16* wr = Wt + (size_t)(1 * H_ + bcol) * E_;
  const __bf16* wh = Wt + (size_t)(2 * H_ + bcol) * E_;
  const __bf16* uz = Ut + (size_t)(0 * H_ + bcol) * H_;
  const __bf16* ur = Ut + (size_t)(1 * H_ + bcol) * H_;
  const __bf16* uh = Ut + (size_t)(2 * H_ + bcol) * H_;

  v8f accZ = {}, accR = {}, accXH = {}, accRH = {};

  // Input-projection contribution (K = 256)
#pragma unroll
  for (int kk = 0; kk < E_ / 32; ++kk) {
    v16bf a = load_ab32(axrow + kk * 32, lane);
    accZ  = wmma_bf16(a, load_ab32(wz + kk * 32, lane), accZ);
    accR  = wmma_bf16(a, load_ab32(wr + kk * 32, lane), accR);
    accXH = wmma_bf16(a, load_ab32(wh + kk * 32, lane), accXH);
  }
  // Recurrent contribution (K = 512)
#pragma unroll
  for (int kk = 0; kk < H_ / 32; ++kk) {
    v16bf a = load_ab32(ahrow + kk * 32, lane);
    accZ  = wmma_bf16(a, load_ab32(uz + kk * 32, lane), accZ);
    accR  = wmma_bf16(a, load_ab32(ur + kk * 32, lane), accR);
    accRH = wmma_bf16(a, load_ab32(uh + kk * 32, lane), accRH);
  }

  const int   n   = bcol;
  const float bz  = bb[n]          + bb[G3_ + n];
  const float br  = bb[H_ + n]     + bb[G3_ + H_ + n];
  const float bxh = bb[2 * H_ + n];
  const float brh = bb[G3_ + 2 * H_ + n];

  const float*  h32p = h32_in  + (size_t)dir * B_ * H_;
  float*        h32n = h32_out + (size_t)dir * B_ * H_;
  __bf16*       h16n = h16_out + (size_t)dir * B_ * H_;

  // C/D layout: VGPR v holds row M = v (lanes 0-15) / v+8 (lanes 16-31), N = lane%16
#pragma unroll
  for (int v = 0; v < 8; ++v) {
    const int   row   = mbase + v + ((lane & 16) ? 8 : 0);
    const float hprev = h32p[(size_t)row * H_ + n];
    const int   sid   = src_ids[row * T_ + t_eff];

    float z = accZ[v] + bz;
    float r = accR[v] + br;
    z = 1.0f / (1.0f + __expf(-z));
    r = 1.0f / (1.0f + __expf(-r));
    const float hh = tanhf(accXH[v] + bxh + r * (accRH[v] + brh));

    float hnew = z * hprev + (1.0f - z) * hh;
    hnew = (sid != 0) ? hnew : hprev;     // mask passthrough (cndmask, no divergence)

    h32n[(size_t)row * H_ + n] = hnew;
    h16n[(size_t)row * H_ + n] = (__bf16)hnew;
    out_seq[((size_t)row * T_ + t_eff) * (2 * H_) + (size_t)dir * H_ + n] = hnew;
    if (t == T_ - 1)
      out_hid[(size_t)row * (2 * H_) + (size_t)dir * H_ + n] = hnew;
  }
}

// ---------------- host ----------------

extern "C" void kernel_launch(void* const* d_in, const int* in_sizes, int n_in,
                              void* d_out, int out_size, void* d_ws, size_t ws_size,
                              hipStream_t stream) {
  const int*   src_ids = (const int*)  d_in[0];
  const float* emb     = (const float*)d_in[1];
  const float* W_f     = (const float*)d_in[2];
  const float* U_f     = (const float*)d_in[3];
  const float* b_f     = (const float*)d_in[4];
  const float* W_b     = (const float*)d_in[5];
  const float* U_b     = (const float*)d_in[6];
  const float* b_b     = (const float*)d_in[7];

  float* out      = (float*)d_out;
  float* out_seq  = out;                                // [B,T,2H]
  float* out_hid  = out + (size_t)B_ * T_ * 2 * H_;     // [B,2H]
  float* out_mask = out_hid + (size_t)B_ * 2 * H_;      // [B,T]

  char*  ws  = (char*)d_ws;
  size_t off = 0;
  auto alloc = [&](size_t bytes) -> char* {
    char* p = ws + off;
    off = (off + bytes + 255) & ~(size_t)255;
    return p;
  };
  __bf16* xbf  = (__bf16*)alloc((size_t)B_ * T_ * E_ * 2);   // 16 MB
  __bf16* Wt_f = (__bf16*)alloc((size_t)G3_ * E_ * 2);
  __bf16* Ut_f = (__bf16*)alloc((size_t)G3_ * H_ * 2);
  __bf16* Wt_b = (__bf16*)alloc((size_t)G3_ * E_ * 2);
  __bf16* Ut_b = (__bf16*)alloc((size_t)G3_ * H_ * 2);
  float*  h32  = (float*) alloc((size_t)4 * B_ * H_ * 4);    // [parity][dir][B*H]
  __bf16* h16  = (__bf16*)alloc((size_t)4 * B_ * H_ * 2);

  const int NSTATE = 4 * B_ * H_;
  init_state<<<(NSTATE + 255) / 256, 256, 0, stream>>>(h32, h16, NSTATE);
  embed_cast<<<B_ * T_, E_, 0, stream>>>(src_ids, emb, xbf, out_mask);
  wcast_t<<<G3_, 256, 0, stream>>>(W_f, Wt_f, E_);
  wcast_t<<<G3_, 256, 0, stream>>>(U_f, Ut_f, H_);
  wcast_t<<<G3_, 256, 0, stream>>>(W_b, Wt_b, E_);
  wcast_t<<<G3_, 256, 0, stream>>>(U_b, Ut_b, H_);

  for (int t = 0; t < T_; ++t) {
    const int pin = t & 1, pout = pin ^ 1;
    gru_step<<<256, 256, 0, stream>>>(
        xbf, Wt_f, Ut_f, Wt_b, Ut_b, b_f, b_b, src_ids,
        h32 + (size_t)pin * 2 * B_ * H_, h32 + (size_t)pout * 2 * B_ * H_,
        h16 + (size_t)pin * 2 * B_ * H_, h16 + (size_t)pout * 2 * B_ * H_,
        out_seq, out_hid, t);
  }
}
